// ConvLocalAutoEncoder_88330297409963
// MI455X (gfx1250) — matrix operand order repaired and checked by reading, not compiled, not run on hardware
//
#include <hip/hip_runtime.h>
#include <hip/hip_bf16.h>

typedef __attribute__((ext_vector_type(16))) _Float16 v16h;
typedef __attribute__((ext_vector_type(8)))  float    v8f;

#define GH 256
#define GW 256
#define EMB 16
#define KS 5
#define HC 252
#define WC 252
#define NPOS (HC*WC)               // 63504
#define NSAMP 5
#define CHUNK 1024
#define NCHUNK ((NPOS + CHUNK - 1) / CHUNK)   // 63
#define CSTRIDE 64                 // per-batch stride for chunk arrays

// ---------------------------------------------------------------------------
// Kernel 1: valid conv (1->16ch) + bias + relu + 1e-4*mean_c(lat^2) -> logits.
// Implicit GEMM via v_wmma_f32_16x16x32_f16 with A = weights (M=16 channels,
// K=25 taps padded to 32) and B = image patches (K x 16 positions). The D
// layout (lane = position, VGPR r = channel 8*half+r) keeps the channel
// reduction in-lane: 8 FMAs + one shfl_xor(16). 4 tiles per wave reuse A.
// ---------------------------------------------------------------------------
__global__ __launch_bounds__(256) void k_conv_logits(
    const float* __restrict__ images, const float* __restrict__ conv_w,
    const float* __restrict__ conv_b, float* __restrict__ logits)
{
  const int b     = blockIdx.y;
  const int wave  = threadIdx.x >> 5;
  const int lane  = threadIdx.x & 31;
  const int half  = lane >> 4;
  const int mrow  = lane & 15;
  const int gwave = blockIdx.x * 8 + wave;
  const int nbase = gwave * 64;
  if (nbase >= NPOS) return;         // wave-uniform (EXEC stays full at WMMA)

  const float* img = images + (size_t)b * (GH * GW);

  // A fragment = conv weights, loop-invariant. ISA 16-bit A layout:
  // lane = M row (channel mrow); half0 holds K 0-7 & 16-23, half1 K 8-15 & 24-31.
  v16h afrag;
  #pragma unroll
  for (int i = 0; i < 16; ++i) {
    int k  = (i < 8) ? (half * 8 + i) : (16 + half * 8 + (i - 8));
    int kk = (k < 25) ? k : 24;
    float v = conv_w[mrow * 25 + kk];
    afrag[i] = (_Float16)((k < 25) ? v : 0.0f);
  }
  float bias[8];
  #pragma unroll
  for (int r = 0; r < 8; ++r) bias[r] = conv_b[8 * half + r];

  #pragma unroll
  for (int t = 0; t < 4; ++t) {
    const int n0 = nbase + t * 16;
    int nm = n0 + mrow; if (nm > NPOS - 1) nm = NPOS - 1;
    const int h = nm / WC, w = nm % WC;

    // B fragment = patch column for position nm. ISA 16-bit B layout:
    // lane = N column (position mrow); half0 holds K 0-7 & 16-23, half1 K 8-15 & 24-31.
    v16h bfrag;
    #pragma unroll
    for (int i = 0; i < 16; ++i) {
      int k  = (i < 8) ? (half * 8 + i) : (16 + half * 8 + (i - 8));
      int kk = (k < 25) ? k : 24;
      float v = img[(h + kk / KS) * GW + (w + kk % KS)];
      bfrag[i] = (_Float16)((k < 25) ? v : 0.0f);
    }

    v8f c = {};
    c = __builtin_amdgcn_wmma_f32_16x16x32_f16(false, afrag, false, bfrag,
                                               (short)0, c, false, false);

    // D: lane holds position N=mrow; VGPR r holds channel M = 8*half + r.
    float ssum = 0.0f;
    #pragma unroll
    for (int r = 0; r < 8; ++r) {
      float tc = c[r] + bias[r];
      tc = tc > 0.0f ? tc : 0.0f;
      ssum = fmaf(tc, tc, ssum);
    }
    ssum += __shfl_xor(ssum, 16, 32);   // combine channel halves
    if (half == 0 && (n0 + mrow) < NPOS)
      logits[(size_t)b * NPOS + n0 + mrow] = 1e-4f * (ssum * (1.0f / EMB));
  }
}

// ---------------------------------------------------------------------------
// Kernel 2a: per-(batch, chunk) sum of exp(logit). Logits are in [0, ~1e-3]
// so softmax max-subtraction is unnecessary (exp in [1, 1.001]).
// ---------------------------------------------------------------------------
__global__ __launch_bounds__(256) void k_chunksum(
    const float* __restrict__ logits, float* __restrict__ chunksum)
{
  const int c = blockIdx.x, b = blockIdx.y, tid = threadIdx.x;
  __shared__ float sm[256];
  const size_t lbase = (size_t)b * NPOS;
  const int base = c * CHUNK + tid * 4;
  float s = 0.0f;
  #pragma unroll
  for (int i = 0; i < 4; ++i) {
    int n = base + i;
    s += (n < NPOS) ? __expf(logits[lbase + n]) : 0.0f;
  }
  sm[tid] = s; __syncthreads();
  for (int off = 128; off > 0; off >>= 1) {
    if (tid < off) sm[tid] += sm[tid + off];
    __syncthreads();
  }
  if (tid == 0) chunksum[b * CSTRIDE + c] = sm[0];
}

// ---------------------------------------------------------------------------
// Kernel 2b: per-batch exclusive scan of the 63 chunk sums + total Z;
// also initializes the idx result slots.
// ---------------------------------------------------------------------------
__global__ __launch_bounds__(64) void k_scan_chunks(
    const float* __restrict__ chunksum, float* __restrict__ chunkoff,
    float* __restrict__ Zout, unsigned* __restrict__ idxOut, int B)
{
  const int b = blockIdx.x, tid = threadIdx.x;
  if (tid == 0) {
    float acc = 0.0f;
    for (int c = 0; c < NCHUNK; ++c) {
      chunkoff[b * CSTRIDE + c] = acc;
      acc += chunksum[b * CSTRIDE + c];
    }
    Zout[b] = acc;
  }
  if (tid < NSAMP) idxOut[tid * B + b] = 0xFFFFFFFFu;
}

// ---------------------------------------------------------------------------
// Kernel 2c: parallel first-hit search. In-chunk inclusive prefix via
// wave shfl_up scan (no barriers) + one cross-wave combine; test the 5 draws
// (u * Z <= unnormalized CDF); wave-min via shfl_xor; one atomicMin per wave.
// ---------------------------------------------------------------------------
__global__ __launch_bounds__(256) void k_find(
    const float* __restrict__ logits, const float* __restrict__ rand_u,
    const float* __restrict__ chunkoff, const float* __restrict__ Zarr,
    unsigned* __restrict__ idxOut, int B)
{
  const int c = blockIdx.x, b = blockIdx.y;
  const int tid = threadIdx.x, lane = tid & 31, waveid = tid >> 5;
  __shared__ float wsum[8];

  const size_t lbase = (size_t)b * NPOS;
  const int base = c * CHUNK + tid * 4;

  float x[4];
  #pragma unroll
  for (int i = 0; i < 4; ++i) {
    int n = base + i;
    x[i] = (n < NPOS) ? __expf(logits[lbase + n]) : 0.0f;
  }
  const float p0 = x[0], p1 = p0 + x[1], p2 = p1 + x[2], p3 = p2 + x[3];
  const float lsum = p3;

  // wave32 inclusive scan of per-thread sums (no barriers)
  float scan = lsum;
  #pragma unroll
  for (int off = 1; off < 32; off <<= 1) {
    float up = __shfl_up(scan, off, 32);
    if (lane >= off) scan += up;
  }
  if (lane == 31) wsum[waveid] = scan;
  __syncthreads();
  float woff = 0.0f;
  #pragma unroll
  for (int j = 0; j < 8; ++j) woff += (j < waveid) ? wsum[j] : 0.0f;

  const float excl = chunkoff[b * CSTRIDE + c] + woff + (scan - lsum);
  const float Z    = Zarr[b];
  const float incl[4] = { excl + p0, excl + p1, excl + p2, excl + p3 };

  #pragma unroll
  for (int s = 0; s < NSAMP; ++s) {
    const float* ru = rand_u + ((size_t)s * B + b) * NPOS;
    unsigned best = 0xFFFFFFFFu;
    #pragma unroll
    for (int i = 0; i < 4; ++i) {
      int n = base + i;
      if (n < NPOS && ru[n] * Z <= incl[i]) {
        unsigned un = (unsigned)n;
        best = best < un ? best : un;
      }
    }
    #pragma unroll
    for (int off = 1; off < 32; off <<= 1) {
      unsigned o = __shfl_xor(best, off, 32);
      best = best < o ? best : o;
    }
    if (lane == 0 && best != 0xFFFFFFFFu)
      atomicMin(&idxOut[s * B + b], best);
  }
}

// ---------------------------------------------------------------------------
// Kernel 3a: zero the output image plane.
// ---------------------------------------------------------------------------
__global__ __launch_bounds__(256) void k_zero(float* __restrict__ out, int n)
{
  int i = blockIdx.x * 256 + threadIdx.x;
  if (i < n) out[i] = 0.0f;
}

// ---------------------------------------------------------------------------
// Kernel 3b: sparse transposed conv. One wave per (draw, batch): recompute the
// fp32 16-channel latent column at the selected site, scatter 25 taps.
// ---------------------------------------------------------------------------
__global__ __launch_bounds__(32) void k_decode(
    const float* __restrict__ images, const float* __restrict__ conv_w,
    const float* __restrict__ conv_b, const float* __restrict__ deconv_w,
    const unsigned* __restrict__ idx, float* __restrict__ out, int B)
{
  const int item = blockIdx.x;     // s*B + b
  const int b    = item % B;
  const int tid  = threadIdx.x;
  unsigned n = idx[item];
  if (n >= NPOS) n = 0u;           // JAX argmax(all-False) -> 0 fallback
  const int h = (int)n / WC, w = (int)n % WC;
  const float* img = images + (size_t)b * (GH * GW);

  __shared__ float lat[EMB];
  if (tid < EMB) {
    float acc = conv_b[tid];
    #pragma unroll
    for (int k = 0; k < 25; ++k)
      acc += img[(h + k / KS) * GW + (w + k % KS)] * conv_w[tid * 25 + k];
    lat[tid] = acc > 0.0f ? acc : 0.0f;
  }
  __syncthreads();
  if (tid < 25) {
    const int kh = tid / 5, kw = tid % 5;
    float v = 0.0f;
    #pragma unroll
    for (int e = 0; e < EMB; ++e) v += lat[e] * deconv_w[e * 25 + kh * 5 + kw];
    atomicAdd(&out[(size_t)b * (GH * GW) + (h + kh) * GW + (w + kw)],
              v * (1.0f / NSAMP));
  }
}

// ---------------------------------------------------------------------------
extern "C" void kernel_launch(void* const* d_in, const int* in_sizes, int n_in,
                              void* d_out, int out_size, void* d_ws, size_t ws_size,
                              hipStream_t stream)
{
  const float* images   = (const float*)d_in[0];
  const float* conv_w   = (const float*)d_in[1];
  const float* conv_b   = (const float*)d_in[2];
  const float* deconv_w = (const float*)d_in[3];
  const float* rand_u   = (const float*)d_in[4];
  const int B = in_sizes[0] / (GH * GW);

  char* ws = (char*)d_ws;
  float*    logits   = (float*)ws;                 ws += (size_t)B * NPOS * sizeof(float);
  float*    chunksum = (float*)ws;                 ws += (size_t)B * CSTRIDE * sizeof(float);
  float*    chunkoff = (float*)ws;                 ws += (size_t)B * CSTRIDE * sizeof(float);
  float*    Zarr     = (float*)ws;                 ws += (size_t)B * sizeof(float);
  unsigned* idx      = (unsigned*)ws;

  // 64 positions per wave -> 993 waves per batch; 8 waves per block -> 125 blocks
  dim3 g1(((NPOS + 63) / 64 + 7) / 8, B);
  k_conv_logits<<<g1, 256, 0, stream>>>(images, conv_w, conv_b, logits);

  dim3 g2(NCHUNK, B);
  k_chunksum<<<g2, 256, 0, stream>>>(logits, chunksum);
  k_scan_chunks<<<B, 64, 0, stream>>>(chunksum, chunkoff, Zarr, idx, B);
  k_find<<<g2, 256, 0, stream>>>(logits, rand_u, chunkoff, Zarr, idx, B);

  const int ntot = B * GH * GW;
  k_zero<<<(ntot + 255) / 256, 256, 0, stream>>>((float*)d_out, ntot);

  k_decode<<<NSAMP * B, 32, 0, stream>>>(images, conv_w, conv_b, deconv_w,
                                         idx, (float*)d_out, B);
}
